// MultiHeadAttention_14010183319965
// MI455X (gfx1250) — matrix operand order repaired and checked by reading, not compiled, not run on hardware
//
#include <hip/hip_runtime.h>

// ---------------------------------------------------------------------------
// Problem constants (from reference): B=4, SX=SY=1024, C1=C2=512, H=16, D=32, W=4
// ---------------------------------------------------------------------------
#define WB    4
#define WSX   1024
#define WSY   1024
#define WC1   512
#define WH    16
#define WD    32
#define NROWX (WB * WSX)        // 4096 query rows
#define HD    (WH * WD)         // 512

typedef __attribute__((ext_vector_type(16))) _Float16 v16h;
typedef __attribute__((ext_vector_type(8)))  _Float16 v8h;
typedef __attribute__((ext_vector_type(8)))  float    v8f;

// ---------------------------------------------------------------------------
// WMMA helpers.
// A-fragment (16x32 f16, row-major source, row stride = lda halfs):
//   lane L: m = L%16, ksel = L/16
//   a[0..7]  = A[m][ksel*16 + i]         -> 16B load at (m*lda + ksel*16)
//   a[8..15] = A[m][ksel*16 + 8 + i]     -> 16B load at (m*lda + ksel*16 + 8)
// B-fragment (32x16, from BT[n][k] row-major, row stride = ldb halfs):
//   lane L: n = L%16, K = ksel*16 + i (i=0..15)  -> contiguous 32B load
// C/D (16x16 f32): lane L holds column n = L%16, VGPR r holds row m = ksel*8 + r
// ---------------------------------------------------------------------------
__device__ __forceinline__ v16h load_a16(const _Float16* p) {
  v8h lo = *(const v8h*)(p);
  v8h hi = *(const v8h*)(p + 8);
  v16h r;
#pragma unroll
  for (int i = 0; i < 8; ++i) { r[i] = lo[i]; r[i + 8] = hi[i]; }
  return r;
}

__device__ __forceinline__ v8f wmma_f16(v16h a, v16h b, v8f c) {
  return __builtin_amdgcn_wmma_f32_16x16x32_f16(false, a, false, b, (short)0, c,
                                                false, false);
}

// ---------------------------------------------------------------------------
// DPP16 cross-lane reductions (wave32; DPP rows are 16 lanes, matching the
// 16-lane column groups of the WMMA C/D layout).  Pure-VALU: no LDS permute,
// co-executes with the WMMA/XDL pipe.
//   0xB1  = quad_perm(1,0,3,2)  : lane ^ 1
//   0x4E  = quad_perm(2,3,0,1)  : lane ^ 2
//   0x141 = row_half_mirror     : lane ^ 7  (combines quads within a group of 8)
//   0x140 = row_mirror          : lane ^ 15 (combines halves of the 16-row)
// ---------------------------------------------------------------------------
template <int CTRL>
__device__ __forceinline__ float dpp_movf(float v) {
  int i = __float_as_int(v);
  return __int_as_float(__builtin_amdgcn_update_dpp(i, i, CTRL, 0xF, 0xF, true));
}

__device__ __forceinline__ float rowmax16(float v) {
  v = fmaxf(v, dpp_movf<0xB1>(v));
  v = fmaxf(v, dpp_movf<0x4E>(v));
  v = fmaxf(v, dpp_movf<0x141>(v));
  v = fmaxf(v, dpp_movf<0x140>(v));
  return v;
}

__device__ __forceinline__ float rowsum16(float v) {
  v += dpp_movf<0xB1>(v);
  v += dpp_movf<0x4E>(v);
  v += dpp_movf<0x141>(v);
  v += dpp_movf<0x140>(v);
  return v;
}

// full wave32 sum: 4 DPP steps within 16-rows + one cross-row shuffle
__device__ __forceinline__ float wavesum32(float v) {
  v = rowsum16(v);
  v += __shfl_xor(v, 16, 32);
  return v;
}

// ---------------------------------------------------------------------------
// LayerNorm (rows of 512) + cast to f16. One wave32 per row, 8 rows per block.
// ---------------------------------------------------------------------------
__global__ __launch_bounds__(256) void ln_cast_kernel(
    const float* __restrict__ x, const float* __restrict__ g,
    const float* __restrict__ be, _Float16* __restrict__ out) {
  int lane = threadIdx.x & 31;
  int wid  = threadIdx.x >> 5;
  int row  = blockIdx.x * 8 + wid;
  const float* xp = x + (size_t)row * WC1;

  float v[16];
  float s = 0.f;
#pragma unroll
  for (int j = 0; j < 16; ++j) { v[j] = xp[lane + j * 32]; s += v[j]; }
  s = wavesum32(s);
  float mu = s * (1.0f / WC1);

  float s2 = 0.f;
#pragma unroll
  for (int j = 0; j < 16; ++j) { float d = v[j] - mu; s2 += d * d; }
  s2 = wavesum32(s2);
  float rstd = rsqrtf(s2 * (1.0f / WC1) + 1e-5f);

  _Float16* op = out + (size_t)row * WC1;
#pragma unroll
  for (int j = 0; j < 16; ++j) {
    int c = lane + j * 32;
    op[c] = (_Float16)((v[j] - mu) * rstd * g[c] + be[c]);
  }
}

// ---------------------------------------------------------------------------
// Transpose + cast f32 -> f16 (batched).  out[b][j][i] = in[b][i][j] * scale
// Used once per launch to build BT operands for all GEMMs (weights only).
// ---------------------------------------------------------------------------
__global__ void transpose_cast_kernel(const float* __restrict__ in,
                                      _Float16* __restrict__ out,
                                      int R, int C, float scale) {
  int bt = blockIdx.y;
  size_t base = (size_t)bt * R * C;
  int idx = blockIdx.x * blockDim.x + threadIdx.x;
  if (idx < R * C) {
    int i = idx / C, j = idx % C;
    out[base + (size_t)j * R + i] = (_Float16)(in[base + idx] * scale);
  }
}

// ---------------------------------------------------------------------------
// Generic WMMA GEMM:  C[row][col] = sum_k A[row][k] * BT[col][k]
// Block = 256 threads = 8 waves (2 x 4); wave tile = 32 x 64 (2x4 WMMA tiles).
// Grid = (M/64, N/256).  Templated epilogue.
// ---------------------------------------------------------------------------
#define EPI_H16   0   // store f16 row-major
#define EPI_VT    1   // scatter to Vt[B][H][D][SY] (f16)
#define EPI_RESID 2   // f32: resid + acc + bias
#define EPI_GELU  3   // f16: exact gelu(acc + bias)

template <int EPI>
__global__ __launch_bounds__(256) void gemm_wmma_kernel(
    const _Float16* __restrict__ A, const _Float16* __restrict__ BT,
    int N, int K, const float* __restrict__ bias,
    const float* __restrict__ resid, void* __restrict__ out) {
  int lane = threadIdx.x & 31;
  int wid  = threadIdx.x >> 5;
  int ks   = lane >> 4;
  int lm   = lane & 15;
  int wm   = wid >> 2, wn = wid & 3;
  int m0 = blockIdx.x * 64 + wm * 32;
  int n0 = blockIdx.y * 256 + wn * 64;

  v8f zero = {};
  v8f acc[2][4];
#pragma unroll
  for (int i = 0; i < 2; ++i)
#pragma unroll
    for (int j = 0; j < 4; ++j) acc[i][j] = zero;

  const _Float16* aptr[2];
  const _Float16* bptr[4];
#pragma unroll
  for (int i = 0; i < 2; ++i)
    aptr[i] = A + (size_t)(m0 + i * 16 + lm) * K + ks * 16;
#pragma unroll
  for (int j = 0; j < 4; ++j)
    bptr[j] = BT + (size_t)(n0 + j * 16 + lm) * K + ks * 16;

  for (int k0 = 0; k0 < K; k0 += 32) {
    // prefetch the streaming A operand one k-tile ahead (global_prefetch_b8)
    if (k0 + 32 < K) {
      __builtin_prefetch(aptr[0] + k0 + 32, 0, 0);
      __builtin_prefetch(aptr[1] + k0 + 32, 0, 0);
    }
    v16h af[2], bf[4];
#pragma unroll
    for (int i = 0; i < 2; ++i) af[i] = load_a16(aptr[i] + k0);
#pragma unroll
    for (int j = 0; j < 4; ++j) bf[j] = *(const v16h*)(bptr[j] + k0);
#pragma unroll
    for (int i = 0; i < 2; ++i)
#pragma unroll
      for (int j = 0; j < 4; ++j) acc[i][j] = wmma_f16(af[i], bf[j], acc[i][j]);
  }

#pragma unroll
  for (int j = 0; j < 4; ++j) {
    int col = n0 + j * 16 + lm;
    float bj = 0.f;
    if constexpr (EPI == EPI_RESID || EPI == EPI_GELU) bj = bias[col];
#pragma unroll
    for (int i = 0; i < 2; ++i)
#pragma unroll
      for (int r = 0; r < 8; ++r) {
        int row = m0 + i * 16 + ks * 8 + r;
        float v = acc[i][j][r];
        if constexpr (EPI == EPI_H16) {
          ((_Float16*)out)[(size_t)row * N + col] = (_Float16)v;
        } else if constexpr (EPI == EPI_VT) {
          int b = row >> 10, s = row & 1023;   // row = b*SY + t
          int h = col >> 5,  d = col & 31;     // col = h*D + d
          ((_Float16*)out)[(((size_t)(b * WH + h)) * WD + d) * WSY + s] =
              (_Float16)v;
        } else if constexpr (EPI == EPI_RESID) {
          ((float*)out)[(size_t)row * N + col] =
              resid[(size_t)row * N + col] + v + bj;
        } else {  // EPI_GELU: exact gelu = 0.5*x*(1+erf(x/sqrt(2)))
          float t = v + bj;
          ((_Float16*)out)[(size_t)row * N + col] =
              (_Float16)(0.5f * t * (1.0f + erff(t * 0.7071067811865475f)));
        }
      }
  }
}

// ---------------------------------------------------------------------------
// Flash attention: one wave handles 16 query rows for one (b,h), streaming
// over SY in steps of 32 keys with online softmax.  Q is pre-scaled by
// log2(e)/sqrt(D) (folded into Wq), so all exponentials are raw exp2f and the
// softmax is mathematically identical to base-e.  Q/K layout: [b*S+s][h*D+d]
// f16; V layout: Vt[b][h][d][SY] f16 so the A*V B-fragment is contiguous.
// P (16x32 f16) round-trips through wave-private LDS to convert the C-matrix
// layout into the A-fragment layout (DS ops are in-order within a wave).
// Softmax row reductions use DPP16 (VALU) and co-execute with the WMMA pipe.
// ---------------------------------------------------------------------------
__global__ __launch_bounds__(256) void attn_kernel(
    const _Float16* __restrict__ Qf, const _Float16* __restrict__ Kf,
    const _Float16* __restrict__ Vt, _Float16* __restrict__ Of) {
  __shared__ __align__(16) _Float16 plds[8 * 512];
  int lane = threadIdx.x & 31;
  int wid  = threadIdx.x >> 5;
  int ks = lane >> 4, lm = lane & 15;
  int bh = blockIdx.x;
  int b = bh >> 4, h = bh & 15;
  int m0 = blockIdx.y * 128 + wid * 16;

  // Q fragment: 16 rows x D=32 (full K-dim of one WMMA)
  const _Float16* qp = Qf + ((size_t)b * WSX + (m0 + lm)) * HD + h * WD + ks * 16;
  v16h aq = load_a16(qp);

  const _Float16* kbase = Kf + ((size_t)b * WSY) * HD + h * WD + ks * 16;
  const _Float16* vbase = Vt + ((size_t)(b * WH + h)) * WD * WSY + ks * 16;

  float mr[8], lr[8];
#pragma unroll
  for (int r = 0; r < 8; ++r) { mr[r] = -INFINITY; lr[r] = 0.f; }
  v8f o0 = {}, o1 = {};

  _Float16* pl = plds + wid * 512;   // wave-private 16x32 f16 tile

  for (int t0 = 0; t0 < WSY; t0 += 32) {
    // S = Q * K^T for 32 keys (two 16-key tiles); S is in log2 units
    v16h bk0 = *(const v16h*)(kbase + (size_t)(t0 + lm) * HD);
    v16h bk1 = *(const v16h*)(kbase + (size_t)(t0 + 16 + lm) * HD);
    v8f zero = {};
    v8f s0 = wmma_f16(aq, bk0, zero);
    v8f s1 = wmma_f16(aq, bk1, zero);

    // online softmax; C-layout row m = ks*8 + r lives in lanes {ks*16..+15}
#pragma unroll
    for (int r = 0; r < 8; ++r) {
      float vmax = rowmax16(fmaxf(s0[r], s1[r]));
      float mn = fmaxf(mr[r], vmax);
      float sc = exp2f(mr[r] - mn);
      float p0 = exp2f(s0[r] - mn);
      float p1 = exp2f(s1[r] - mn);
      float rs = rowsum16(p0 + p1);
      lr[r] = lr[r] * sc + rs;
      o0[r] *= sc;
      o1[r] *= sc;
      mr[r] = mn;
      int rowoff = (ks * 8 + r) * 32;        // row-major 16x32 P tile in LDS
      pl[rowoff + lm]      = (_Float16)p0;
      pl[rowoff + 16 + lm] = (_Float16)p1;
    }

    // P back as A-fragment (DS ops are in-order within a wave)
    v16h ap = load_a16(pl + lm * 32 + ks * 16);

    // O += P * V  (two d-halves of D=32)
    v16h bv0 = *(const v16h*)(vbase + (size_t)lm * WSY + t0);
    v16h bv1 = *(const v16h*)(vbase + (size_t)(16 + lm) * WSY + t0);
    o0 = wmma_f16(ap, bv0, o0);
    o1 = wmma_f16(ap, bv1, o1);
  }

#pragma unroll
  for (int r = 0; r < 8; ++r) {
    float inv = 1.0f / lr[r];
    int mrow = m0 + ks * 8 + r;
    _Float16* op = Of + ((size_t)b * WSX + mrow) * HD + h * WD;
    op[lm]      = (_Float16)(o0[r] * inv);
    op[16 + lm] = (_Float16)(o1[r] * inv);
  }
}

// ---------------------------------------------------------------------------
// Launcher
// ---------------------------------------------------------------------------
extern "C" void kernel_launch(void* const* d_in, const int* in_sizes, int n_in,
                              void* d_out, int out_size, void* d_ws,
                              size_t ws_size, hipStream_t stream) {
  (void)in_sizes; (void)n_in; (void)out_size; (void)ws_size;
  const float* x    = (const float*)d_in[0];
  const float* y    = (const float*)d_in[1];
  const float* ln1g = (const float*)d_in[2];
  const float* ln1b = (const float*)d_in[3];
  const float* ln2g = (const float*)d_in[4];
  const float* ln2b = (const float*)d_in[5];
  const float* Wq   = (const float*)d_in[6];
  const float* Wk   = (const float*)d_in[7];
  const float* Wv   = (const float*)d_in[8];
  const float* W1   = (const float*)d_in[9];
  const float* b1   = (const float*)d_in[10];
  const float* ln3g = (const float*)d_in[11];
  const float* ln3b = (const float*)d_in[12];
  const float* W2   = (const float*)d_in[13];
  const float* b2   = (const float*)d_in[14];
  const float* W3   = (const float*)d_in[15];
  const float* b3   = (const float*)d_in[16];
  float* outp = (float*)d_out;

  // workspace carve (~58 MB total), 256B aligned slices
  char* w = (char*)d_ws;
  auto take = [&](size_t bytes) -> void* {
    void* p = (void*)w;
    w += (bytes + 255) & ~(size_t)255;
    return p;
  };
  _Float16* xn   = (_Float16*)take((size_t)NROWX * WC1 * 2);
  _Float16* yn   = (_Float16*)take((size_t)NROWX * WC1 * 2);
  _Float16* Qh   = (_Float16*)take((size_t)NROWX * HD * 2);
  _Float16* Kh   = (_Float16*)take((size_t)NROWX * HD * 2);
  _Float16* Vth  = (_Float16*)take((size_t)WB * WH * WD * WSY * 2);
  _Float16* Oh   = (_Float16*)take((size_t)NROWX * HD * 2);
  float*    xout = (float*)   take((size_t)NROWX * WC1 * 4);
  _Float16* fn   = (_Float16*)take((size_t)NROWX * WC1 * 2);
  _Float16* gh   = (_Float16*)take((size_t)NROWX * (WC1 * 4) * 2);
  _Float16* WqT  = (_Float16*)take((size_t)WH * WD * WC1 * 2);
  _Float16* WkT  = (_Float16*)take((size_t)WH * WD * WC1 * 2);
  _Float16* WvT  = (_Float16*)take((size_t)WH * WD * WC1 * 2);
  _Float16* W1T  = (_Float16*)take((size_t)HD * WC1 * 2);
  _Float16* W2T  = (_Float16*)take((size_t)WC1 * (WC1 * 4) * 2);
  _Float16* W3T  = (_Float16*)take((size_t)(WC1 * 4) * WC1 * 2);

  // 1) LayerNorms -> f16
  ln_cast_kernel<<<NROWX / 8, 256, 0, stream>>>(x, ln1g, ln1b, xn);
  ln_cast_kernel<<<NROWX / 8, 256, 0, stream>>>(y, ln2g, ln2b, yn);

  // 2) Weight transposes -> BT f16.  Fold softmax scale (1/sqrt(D)) AND
  //    log2(e) into WqT so attention uses raw exp2.
  const float qscale = 0.17677669529663687f * 1.4426950408889634f;
  transpose_cast_kernel<<<dim3((WC1 * WD + 255) / 256, WH), 256, 0, stream>>>(
      Wq, WqT, WC1, WD, qscale);
  transpose_cast_kernel<<<dim3((WC1 * WD + 255) / 256, WH), 256, 0, stream>>>(
      Wk, WkT, WC1, WD, 1.0f);
  transpose_cast_kernel<<<dim3((WC1 * WD + 255) / 256, WH), 256, 0, stream>>>(
      Wv, WvT, WC1, WD, 1.0f);
  transpose_cast_kernel<<<dim3((HD * WC1 + 255) / 256, 1), 256, 0, stream>>>(
      W1, W1T, HD, WC1, 1.0f);
  transpose_cast_kernel<<<dim3((WC1 * WC1 * 4 + 255) / 256, 1), 256, 0, stream>>>(
      W2, W2T, WC1, WC1 * 4, 1.0f);
  transpose_cast_kernel<<<dim3((WC1 * 4 * WC1 + 255) / 256, 1), 256, 0, stream>>>(
      W3, W3T, WC1 * 4, WC1, 1.0f);

  // 3) QKV projections (4096x512x512 each)
  dim3 g512(NROWX / 64, WC1 / 256);
  gemm_wmma_kernel<EPI_H16><<<g512, 256, 0, stream>>>(xn, WqT, HD, WC1, nullptr,
                                                      nullptr, Qh);
  gemm_wmma_kernel<EPI_H16><<<g512, 256, 0, stream>>>(yn, WkT, HD, WC1, nullptr,
                                                      nullptr, Kh);
  gemm_wmma_kernel<EPI_VT><<<g512, 256, 0, stream>>>(yn, WvT, HD, WC1, nullptr,
                                                     nullptr, Vth);

  // 4) Flash attention
  attn_kernel<<<dim3(WB * WH, WSX / 128), 256, 0, stream>>>(Qh, Kh, Vth, Oh);

  // 5) x_out = x + O @ W1 + b1
  gemm_wmma_kernel<EPI_RESID><<<g512, 256, 0, stream>>>(Oh, W1T, WC1, HD, b1, x,
                                                        xout);

  // 6) f = LN3(x_out)
  ln_cast_kernel<<<NROWX / 8, 256, 0, stream>>>(xout, ln3g, ln3b, fn);

  // 7) g = gelu(f @ W2 + b2)   (4096 x 2048 x 512)
  gemm_wmma_kernel<EPI_GELU><<<dim3(NROWX / 64, (WC1 * 4) / 256), 256, 0,
                               stream>>>(fn, W2T, WC1 * 4, WC1, b2, nullptr, gh);

  // 8) out = x_out + g @ W3 + b3   (4096 x 512 x 2048)
  gemm_wmma_kernel<EPI_RESID><<<g512, 256, 0, stream>>>(gh, W3T, WC1, WC1 * 4,
                                                        b3, xout, outp);
}
